// Deformation_4243427688852
// MI455X (gfx1250) — matrix operand order repaired and checked by reading, not compiled, not run on hardware
//
#include <hip/hip_runtime.h>
#include <hip/hip_bf16.h>
#include <math.h>

typedef __attribute__((ext_vector_type(16))) __bf16 v16bf;
typedef __attribute__((ext_vector_type(8)))  float  v8f;
typedef __attribute__((ext_vector_type(4)))  unsigned int v4u;
typedef __attribute__((ext_vector_type(8)))  int  v8i;
typedef __attribute__((ext_vector_type(4)))  int  v4i;

#define TILE_M   128          // rows per workgroup (8 waves x 16 rows)
#define KP       256          // padded contraction dim (255 -> 256)
#define LDSTRIDE 264          // padded row stride (bf16 elems): 528 B rows, bank-conflict free
#define W1T_ELEMS_PER_HEAD (256*256)
#define W2T_TOTAL_ROWS 112    // 16+16+16+16+48 padded output rows

// LDS layout (bytes)
#define LDS_W1    0
#define LDS_W1_SZ (256 * LDSTRIDE * 2)          // 135168
#define LDS_HR    (LDS_W1 + LDS_W1_SZ)
#define LDS_HR_SZ (TILE_M * LDSTRIDE * 2)       // 67584
#define LDS_H1    (LDS_HR + LDS_HR_SZ)
#define LDS_H1_SZ (TILE_M * LDSTRIDE * 2)       // 67584
#define LDS_MASK  (LDS_H1 + LDS_H1_SZ)
#define LDS_TOTAL (LDS_MASK + TILE_M * 4)

// ---------------- fragment loads ----------------
// A (16x32 bf16): lane l: m = l&15, h = l>>4 ; v[i] = K(kb+16h+i) -> 16 contiguous bf16
__device__ __forceinline__ v16bf load_frag_a(const __bf16* p) {
    union { uint4 u[2]; v16bf v; } x;
    x.u[0] = *(const uint4*)(p);
    x.u[1] = *(const uint4*)(p + 8);
    return x.v;
}
// B (32x16 bf16) from W^T[n][k]: lane l: n = l&15, h = l>>4 ; v[i] = B[K=16h+i][n] -> 16 contiguous bf16
__device__ __forceinline__ v16bf load_frag_b(const __bf16* p) {
    union { uint4 u[2]; v16bf v; } x;
    x.u[0] = *(const uint4*)(p);
    x.u[1] = *(const uint4*)(p + 8);
    return x.v;
}

__device__ __forceinline__ v8f wmma_bf16(v16bf a, v16bf b, v8f c) {
    return __builtin_amdgcn_wmma_f32_16x16x32_bf16(false, a, false, b, (short)0, c, false, false);
}

// ---------------- TDM: stage one head's W1 (128 KB) into LDS with padded rows ----------------
// Tensor: 16384 x 8B elements, 1 row. LDS padding: after every 512 B insert 16 B
// (pad_interval code 6 = 128 dwords, pad_amount code 3 = 4 dwords) -> 528 B rows = LDSTRIDE.
__device__ __forceinline__ void tdm_load_w1(const __bf16* gsrc, unsigned int lds_byte_off) {
    unsigned long long ga = (unsigned long long)(__UINTPTR_TYPE__)gsrc;
    v4u g0 = { 1u,                                   // count = 1 (valid descriptor)
               lds_byte_off,                         // lds_addr
               (unsigned int)ga,                     // global_addr[31:0]
               (unsigned int)((ga >> 32) & 0x01FFFFFFu) | (2u << 30) };  // addr[56:32] | type=2
    v8i g1 = { (int)(0x30000u | (1u << 20) | (6u << 22) | (3u << 25)),  // data_size=8B, pad en/intvl/amt
               (int)(16384u << 16),                  // tensor_dim0[15:0] in [31:16]
               (int)(1u << 16),                      // tensor_dim0 hi=0, tensor_dim1=1
               (int)(16384u << 16),                  // tensor_dim1 hi=0, tile_dim0=16384
               1,                                    // tile_dim1=1, tile_dim2=0
               16384,                                // tensor_dim0_stride[31:0]
               0, 0 };
    v4i z4 = { 0, 0, 0, 0 };
    v8i z8 = { 0, 0, 0, 0, 0, 0, 0, 0 };
    __builtin_amdgcn_tensor_load_to_lds(g0, g1, z4, z4, z8, 0);
}

// ---------------- weight prep: fp32 -> bf16, transposed [n][k], zero padded ----------------
__global__ void __launch_bounds__(256)
prep_weights(const float* w1_0, const float* w1_1, const float* w1_2, const float* w1_3, const float* w1_4,
             const float* w2_0, const float* w2_1, const float* w2_2, const float* w2_3, const float* w2_4,
             __bf16* ws, int total)
{
    int idx = blockIdx.x * 256 + threadIdx.x;
    if (idx >= total) return;
    if (idx < 5 * W1T_ELEMS_PER_HEAD) {
        int h   = idx >> 16;
        int rem = idx & 65535;
        int nn  = rem >> 8;
        int k   = rem & 255;
        const float* w1 = (h==0)?w1_0:(h==1)?w1_1:(h==2)?w1_2:(h==3)?w1_3:w1_4;
        float v = (nn < 255 && k < 255) ? w1[k*255 + nn] : 0.f;
        ws[idx] = (__bf16)v;
    } else {
        int j      = idx - 5 * W1T_ELEMS_PER_HEAD;
        int rowIdx = j >> 8;        // 0..111
        int k      = j & 255;
        int nbase, od; const float* w2;
        if      (rowIdx < 16) { nbase = 0;  od = 1;  w2 = w2_0; }
        else if (rowIdx < 32) { nbase = 16; od = 3;  w2 = w2_1; }
        else if (rowIdx < 48) { nbase = 32; od = 3;  w2 = w2_2; }
        else if (rowIdx < 64) { nbase = 48; od = 4;  w2 = w2_3; }
        else                  { nbase = 64; od = 48; w2 = w2_4; }
        int nn = rowIdx - nbase;
        float v = (nn < od && k < 255) ? w2[k*od + nn] : 0.f;
        ws[idx] = (__bf16)v;
    }
}

// ---------------- one MLP head, fully const-folded ----------------
// H: head id; NT2: # of 16-wide out tiles; W2ROW0: row offset into W2T; OUTD: true out dim;
// NEXTH: head to prefetch via TDM during layer 2 (-1 = none)
template<int H, int NT2, int W2ROW0, int OUTD, int NEXTH>
__device__ __forceinline__ void head_pass(
    const __bf16* __restrict__ wbuf, const __bf16* W1lds, const v16bf (&aH)[8],
    __bf16* __restrict__ H1w, float* __restrict__ maskW,
    const float* __restrict__ b1, const float* __restrict__ b2,
    const float* __restrict__ point, const float* __restrict__ scl_in,
    const float* __restrict__ rot_in, const float* __restrict__ shs_emb,
    float* __restrict__ out, int n, int row0w, int lane, int wave)
{
    const int half = lane >> 4;
    const int nl   = lane & 15;

    // ---- Layer 1: H1 = relu(Hr @ W1 + b1), B fragments from LDS-staged W1 ----
    #pragma unroll 4
    for (int nt = 0; nt < 16; ++nt) {
        v8f acc = {0.f,0.f,0.f,0.f,0.f,0.f,0.f,0.f};
        const __bf16* Bb = W1lds + (nt * 16 + nl) * LDSTRIDE + half * 16;
        #pragma unroll
        for (int kt = 0; kt < 8; ++kt)
            acc = wmma_bf16(aH[kt], load_frag_b(Bb + kt * 32), acc);
        int ng = nt * 16 + nl;
        float bias = (ng < 255) ? b1[ng] : 0.f;
        #pragma unroll
        for (int r = 0; r < 8; ++r) {
            float vv = fmaxf(acc[r] + bias, 0.f);
            if (ng >= 255) vv = 0.f;
            H1w[(r + 8 * half) * LDSTRIDE + ng] = (__bf16)vv;  // wave-local, in-order LDS
        }
    }

    // all waves finished reading W1lds -> safe to overwrite with next head's W1
    __syncthreads();
    if constexpr (NEXTH >= 0) {
        if (wave == 0)
            tdm_load_w1(wbuf + NEXTH * W1T_ELEMS_PER_HEAD, (unsigned int)LDS_W1);
    }

    // ---- hoist A fragments of H1 (loaded once per head) ----
    v16bf a1[8];
    const __bf16* Ab = H1w + nl * LDSTRIDE + half * 16;
    #pragma unroll
    for (int kt = 0; kt < 8; ++kt)
        a1[kt] = load_frag_a(Ab + kt * 32);

    // ---- Layer 2: out = H1 @ W2 + b2 (W2 from global, tiny & cached), fused epilogue ----
    #pragma unroll
    for (int o = 0; o < NT2; ++o) {
        v8f acc = {0.f,0.f,0.f,0.f,0.f,0.f,0.f,0.f};
        const __bf16* Bb = wbuf + 5 * W1T_ELEMS_PER_HEAD
                         + (W2ROW0 + o * 16 + nl) * KP + half * 16;
        #pragma unroll
        for (int kt = 0; kt < 8; ++kt)
            acc = wmma_bf16(a1[kt], load_frag_b(Bb + kt * 32), acc);

        int c = o * 16 + nl;
        if constexpr (H == 0) {
            if (nl == 0) {
                float b2v = b2[0];
                #pragma unroll
                for (int r = 0; r < 8; ++r) {
                    float val = acc[r] + b2v;
                    maskW[r + 8 * half] = 1.f / (1.f + __expf(-val));
                }
            }
        } else {
            if (c < OUTD) {
                float b2v = b2[c];
                #pragma unroll
                for (int r = 0; r < 8; ++r) {
                    int m   = r + 8 * half;
                    int row = row0w + m;
                    if (row < n) {
                        float val = acc[r] + b2v;
                        float mk  = maskW[m];
                        if constexpr (H == 1) {          // pos: pts + dx_out
                            out[(size_t)row * 3 + c] = point[row*3 + c] + val * mk;
                            out[(size_t)59 * n + (size_t)row * 3 + c] = val;
                        } else if constexpr (H == 2) {   // scales
                            out[(size_t)3 * n + (size_t)row * 3 + c] = scl_in[row*3 + c] + val * mk;
                        } else if constexpr (H == 3) {   // rotations
                            out[(size_t)6 * n + (size_t)row * 4 + c] = rot_in[row*4 + c] + val * mk;
                        } else {                         // shs + dshs
                            out[(size_t)11 * n + (size_t)row * 48 + c] = shs_emb[row*48 + c] + val * mk;
                            out[(size_t)62 * n + (size_t)row * 48 + c] = val;
                        }
                    }
                }
            }
        }
    }

    // publish next head's W1 (TDM overlapped with layer 2 above)
    if constexpr (NEXTH >= 0) {
        if (wave == 0) __builtin_amdgcn_s_wait_tensorcnt(0);
        __syncthreads();
    }
}

// ---------------- fused deformation MLP ----------------
__global__ void __launch_bounds__(256)
deform_kernel(const float* __restrict__ opacity, const float* __restrict__ shs_emb,
              const float* __restrict__ time_emb, const float* __restrict__ sem,
              const float* __restrict__ point,   const float* __restrict__ scl_in,
              const float* __restrict__ rot_in,  const float* __restrict__ dx,
              const float* __restrict__ b1m, const float* __restrict__ b2m,
              const float* __restrict__ b1p, const float* __restrict__ b2p,
              const float* __restrict__ b1s, const float* __restrict__ b2s,
              const float* __restrict__ b1r, const float* __restrict__ b2r,
              const float* __restrict__ b1h, const float* __restrict__ b2h,
              const __bf16* __restrict__ wbuf, float* __restrict__ out, int n)
{
    extern __shared__ char smem[];
    __bf16* W1lds = (__bf16*)(smem + LDS_W1);   // 256 x LDSTRIDE (TDM-staged, padded rows)
    __bf16* Hr    = (__bf16*)(smem + LDS_HR);   // 128 x LDSTRIDE
    __bf16* H1    = (__bf16*)(smem + LDS_H1);   // 8 waves x 16 x LDSTRIDE
    float*  maskL = (float*)(smem + LDS_MASK);  // 128 floats

    const int tid     = threadIdx.x;
    const int lane    = tid & 31;
    const int wave    = tid >> 5;
    const int wg_row0 = blockIdx.x * TILE_M;

    // kick off TDM for head 0's W1 -> overlaps with the hidden-vector build below
    if (wave == 0)
        tdm_load_w1(wbuf, (unsigned int)LDS_W1);

    const float t = time_emb[0];

    // ---- Phase A: build relu(hidden) in bf16, zero-padded col 255 ----
    for (int i = tid; i < TILE_M * KP; i += 256) {
        int r = i >> 8;
        int c = i & 255;
        int row = wg_row0 + r;
        float v = 0.f;
        if (row < n) {
            if      (c < 3)   v = point[row*3 + c];
            else if (c < 7)   v = rot_in[row*4 + (c-3)];
            else if (c < 10)  v = scl_in[row*3 + (c-7)];
            else if (c == 10) v = opacity[row];
            else if (c < 59)  v = shs_emb[row*48 + (c-11)];
            else if (c < 187) v = sem[row*128 + (c-59)];
            else if (c < 190) v = dx[row*3 + (c-187)];
            else if (c == 190) v = time_emb[row];
            else if (c < 255) {
                int e = c - 191;                 // 0..63
                int p = e >> 1;                  // 0..31
                float freq = __powf(10000.f, (float)(2*p) * (1.f/32.f));
                float a = t / freq;
                v = (e & 1) ? __cosf(a) : __sinf(a);
            }
            v = fmaxf(v, 0.f);
        }
        Hr[r * LDSTRIDE + c] = (__bf16)v;
    }
    // opacity passthrough output
    if (tid < TILE_M) {
        int row = wg_row0 + tid;
        if (row < n) out[(size_t)10 * n + row] = opacity[row];
    }
    // W1(head0) staged + Hr built -> publish both
    if (wave == 0) __builtin_amdgcn_s_wait_tensorcnt(0);
    __syncthreads();

    const int half  = lane >> 4;
    const int nl    = lane & 15;
    const int row0w = wg_row0 + wave * 16;
    __bf16* H1w   = H1 + wave * 16 * LDSTRIDE;
    float*  maskW = maskL + wave * 16;

    // ---- hoist A fragments of Hr: invariant across all N-tiles and heads ----
    v16bf aH[8];
    const __bf16* Ab = Hr + (wave * 16 + nl) * LDSTRIDE + half * 16;
    #pragma unroll
    for (int kt = 0; kt < 8; ++kt)
        aH[kt] = load_frag_a(Ab + kt * 32);

    head_pass<0, 1,  0, 1,  1>(wbuf, W1lds, aH, H1w, maskW, b1m, b2m, point, scl_in, rot_in, shs_emb, out, n, row0w, lane, wave);
    head_pass<1, 1, 16, 3,  2>(wbuf, W1lds, aH, H1w, maskW, b1p, b2p, point, scl_in, rot_in, shs_emb, out, n, row0w, lane, wave);
    head_pass<2, 1, 32, 3,  3>(wbuf, W1lds, aH, H1w, maskW, b1s, b2s, point, scl_in, rot_in, shs_emb, out, n, row0w, lane, wave);
    head_pass<3, 1, 48, 4,  4>(wbuf, W1lds, aH, H1w, maskW, b1r, b2r, point, scl_in, rot_in, shs_emb, out, n, row0w, lane, wave);
    head_pass<4, 3, 64, 48, -1>(wbuf, W1lds, aH, H1w, maskW, b1h, b2h, point, scl_in, rot_in, shs_emb, out, n, row0w, lane, wave);
}

extern "C" void kernel_launch(void* const* d_in, const int* in_sizes, int n_in,
                              void* d_out, int out_size, void* d_ws, size_t ws_size,
                              hipStream_t stream) {
    const int n = in_sizes[0];   // opacity_emb is (N,1)
    __bf16* ws = (__bf16*)d_ws;  // needs 5*65536 + 112*256 bf16 = ~713 KB

    const int totalPrep = 5 * W1T_ELEMS_PER_HEAD + W2T_TOTAL_ROWS * 256;
    prep_weights<<<(totalPrep + 255) / 256, 256, 0, stream>>>(
        (const float*)d_in[8],  (const float*)d_in[12], (const float*)d_in[16],
        (const float*)d_in[20], (const float*)d_in[24],
        (const float*)d_in[10], (const float*)d_in[14], (const float*)d_in[18],
        (const float*)d_in[22], (const float*)d_in[26],
        ws, totalPrep);

    const size_t smem = (size_t)LDS_TOTAL;   // ~265 KB dynamic LDS (<= 320 KB/WGP)
    const int grid = (n + TILE_M - 1) / TILE_M;
    deform_kernel<<<grid, 256, smem, stream>>>(
        (const float*)d_in[0], (const float*)d_in[1], (const float*)d_in[2],
        (const float*)d_in[3], (const float*)d_in[4], (const float*)d_in[5],
        (const float*)d_in[6], (const float*)d_in[7],
        (const float*)d_in[9],  (const float*)d_in[11],
        (const float*)d_in[13], (const float*)d_in[15],
        (const float*)d_in[17], (const float*)d_in[19],
        (const float*)d_in[21], (const float*)d_in[23],
        (const float*)d_in[25], (const float*)d_in[27],
        ws, (float*)d_out, n);
}
